// IPA_48756468744630
// MI455X (gfx1250) — matrix operand order repaired and checked by reading, not compiled, not run on hardware
//
#include <hip/hip_runtime.h>

// ---------------------------------------------------------------------------
// Types for CDNA5 WMMA / TDM
// ---------------------------------------------------------------------------
typedef __bf16 bf16;
typedef __attribute__((ext_vector_type(16))) __bf16 v16bf;
typedef __attribute__((ext_vector_type(8)))  float  v8f;
typedef __attribute__((ext_vector_type(2)))  float  v2f;
typedef __attribute__((ext_vector_type(4)))  unsigned int v4u;
typedef __attribute__((ext_vector_type(8)))  int    v8i;
typedef __attribute__((ext_vector_type(4)))  int    v4i;

__device__ __forceinline__ bf16 f2bf(float f) {
    unsigned u = __builtin_bit_cast(unsigned, f);
    unsigned r = (u + 0x7fffu + ((u >> 16) & 1u)) >> 16;
    unsigned short s = (unsigned short)r;
    return __builtin_bit_cast(bf16, s);
}

// Problem constants
#define NN 512
#define NODE_DIM 384
#define PAIR_DIM 128
#define HH 12
#define DQ 16
#define DQP 4
#define DV 16
#define DVP 8
#define QKV_DIM 1152   // sum(SIZES)
#define FEAT_DIM 2112  // H*(PAIR_DIM+DV+4*DVP)

// ---------------------------------------------------------------------------
// TDM: issue an async DMA of a 32x128 f32 tile (row-major, stride 128) from
// global memory into LDS. Descriptor per CDNA5 ISA §8.3/8.4. 2-D tensor,
// groups 2/3 (and the trailing group) zeroed. Uniform operands (one issue per
// executing wave). 6-arg builtin form (clang-23 / therock-10.0 headers).
// ---------------------------------------------------------------------------
__device__ __forceinline__ void tdm_load_tile(unsigned long long gaddr, unsigned lds_addr) {
    v4u g0;
    g0[0] = 1u;                                              // count=1, user desc
    g0[1] = lds_addr;                                        // LDS byte address
    g0[2] = (unsigned)(gaddr & 0xffffffffull);               // global_addr[31:0]
    g0[3] = (unsigned)((gaddr >> 32) & 0x01ffffffull)        // global_addr[56:32]
          | (2u << 30);                                      // type = 2 ("image")
    v8i g1;
    g1[0] = (int)(2u << 16);                                 // data_size = 4B
    g1[1] = (int)((unsigned)(PAIR_DIM & 0xffff) << 16);      // tensor_dim0 lo16
    g1[2] = (int)((unsigned)(32 & 0xffff) << 16);            // dim0 hi16=0 | tensor_dim1 lo16
    g1[3] = (int)((unsigned)PAIR_DIM << 16);                 // dim1 hi16=0 | tile_dim0=128
    g1[4] = 32;                                              // tile_dim1=32, tile_dim2=0
    g1[5] = PAIR_DIM;                                        // tensor_dim0_stride lo32
    g1[6] = 0;                                               // stride0 hi | stride1 lo
    g1[7] = 0;                                               // stride1 hi
    v4i g2 = {0, 0, 0, 0};
    v4i g3 = {0, 0, 0, 0};
    v8i g4 = {0, 0, 0, 0, 0, 0, 0, 0};
    __builtin_amdgcn_tensor_load_to_lds(g0, g1, g2, g3, g4, 0);
}

// ---------------------------------------------------------------------------
// K0: convert weights to transposed bf16 (so B fragments read contiguous k)
// ---------------------------------------------------------------------------
__global__ void convw_kernel(bf16* __restrict__ WqkvT, bf16* __restrict__ WoutT,
                             const float* __restrict__ Wqkv, const float* __restrict__ Wout) {
    int idx = blockIdx.x * blockDim.x + threadIdx.x;
    if (idx < NODE_DIM * QKV_DIM) {
        int k = idx / QKV_DIM, n = idx % QKV_DIM;
        WqkvT[(size_t)n * NODE_DIM + k] = f2bf(Wqkv[idx]);
    }
    if (idx < FEAT_DIM * NODE_DIM) {
        int k = idx / NODE_DIM, n = idx % NODE_DIM;
        WoutT[(size_t)n * FEAT_DIM + k] = f2bf(Wout[idx]);
    }
}

// ---------------------------------------------------------------------------
// K1: layernorm of node feats -> bf16
// ---------------------------------------------------------------------------
__global__ void lnnode_kernel(bf16* __restrict__ X, const float* __restrict__ NF,
                              const float* __restrict__ g, const float* __restrict__ b) {
    int i = blockIdx.x;          // 512 blocks
    int tid = threadIdx.x;       // 128 threads
    const float* row = NF + (size_t)i * NODE_DIM;
    float x[3]; float s = 0.f, s2 = 0.f;
#pragma unroll
    for (int e = 0; e < 3; ++e) { x[e] = row[tid + 128 * e]; s += x[e]; s2 += x[e] * x[e]; }
#pragma unroll
    for (int m = 16; m; m >>= 1) { s += __shfl_xor(s, m, 32); s2 += __shfl_xor(s2, m, 32); }
    __shared__ float ws1[4], ws2[4];
    int w = tid >> 5, lane = tid & 31;
    if (lane == 0) { ws1[w] = s; ws2[w] = s2; }
    __syncthreads();
    s  = ws1[0] + ws1[1] + ws1[2] + ws1[3];
    s2 = ws2[0] + ws2[1] + ws2[2] + ws2[3];
    float mean = s / (float)NODE_DIM;
    float rstd = rsqrtf(s2 / (float)NODE_DIM - mean * mean + 1e-5f);
#pragma unroll
    for (int e = 0; e < 3; ++e) {
        int d = tid + 128 * e;
        X[(size_t)i * NODE_DIM + d] = f2bf((x[e] - mean) * rstd * g[d] + b[d]);
    }
}

// ---------------------------------------------------------------------------
// Generic bf16 WMMA GEMM: C[MxN] = A[MxK] * Bt[NxK]^T (+bias), fp32 accum.
// One wave computes one 16x16 tile. M,N mult of 16; K mult of 32.
// ---------------------------------------------------------------------------
__global__ void gemm_bf16_kernel(float* __restrict__ C, const bf16* __restrict__ A,
                                 const bf16* __restrict__ Bt, const float* __restrict__ bias,
                                 int M, int N, int K, int lda, int ldb, int ldc) {
    int wave = blockIdx.x * (blockDim.x >> 5) + (threadIdx.x >> 5);
    int tilesN = N >> 4;
    int totalTiles = (M >> 4) * tilesN;
    if (wave >= totalTiles) return;
    int tm = wave / tilesN, tn = wave % tilesN;
    int row0 = tm << 4, col0 = tn << 4;
    int lane = threadIdx.x & 31;
    int mr = lane & 15, half = lane >> 4;
    v8f c = {};
    for (int k0 = 0; k0 < K; k0 += 32) {
        v16bf a, bb;
        const bf16* pa = A  + (size_t)(row0 + mr) * lda + k0 + 8 * half;
        const bf16* pb = Bt + (size_t)(col0 + mr) * ldb + k0 + 8 * half;
#pragma unroll
        for (int r = 0; r < 8; ++r) {
            int ka = (r < 4) ? 2 * r : 2 * r + 8;
            a[2 * r]     = pa[ka];
            a[2 * r + 1] = pa[ka + 1];
            bb[2 * r]     = pb[ka];
            bb[2 * r + 1] = pb[ka + 1];
        }
        c = __builtin_amdgcn_wmma_f32_16x16x32_bf16(false, a, false, bb, (short)0, c, false, false);
    }
#pragma unroll
    for (int r = 0; r < 8; ++r) {
        int row = row0 + r + 8 * half;
        int col = col0 + mr;
        float v = c[r];
        if (bias) v += bias[col];
        C[(size_t)row * ldc + col] = v;
    }
}

// ---------------------------------------------------------------------------
// K3: per (i,h): rigid transforms; build fused logit operands (K=32 fp32)
//     and transposed bf16 value matrix [h][48][N] (v_s | v_p | pad)
// ---------------------------------------------------------------------------
__global__ void prep_kernel(float* __restrict__ Acat, float* __restrict__ Bcat,
                            bf16* __restrict__ VT, const float* __restrict__ qkv,
                            const float* __restrict__ Rm, const float* __restrict__ Tr,
                            const float* __restrict__ PW) {
    int idx = blockIdx.x * blockDim.x + threadIdx.x;
    if (idx >= NN * HH) return;
    int i = idx / HH, h = idx % HH;
    const float* row = qkv + (size_t)i * QKV_DIM;
    const float* R = Rm + (size_t)i * 9;   // [r][c]
    const float* t = Tr + (size_t)i * 3;
    float pw = log1pf(__expf(PW[h]));                     // softplus
    const float SS = rsqrtf(3.f * (float)DQ);             // scalar scale
    const float PS = rsqrtf(3.f * (float)DQP * 9.f * 1.41421356237309515f);
    float* Ar = Acat + ((size_t)h * NN + i) * 32;
    float* Br = Bcat + ((size_t)h * NN + i) * 32;
#pragma unroll
    for (int d = 0; d < DQ; ++d) {
        Ar[d] = SS * row[h * DQ + d];
        Br[d] = row[192 + h * DQ + d];
    }
    float qsq = 0.f, ksq = 0.f;
#pragma unroll
    for (int p = 0; p < DQP; ++p) {
        const float* qp = row + 576 + (h * DQP + p) * 3;
        const float* kp = row + 720 + (h * DQP + p) * 3;
#pragma unroll
        for (int r = 0; r < 3; ++r) {
            float qg = R[r*3+0]*qp[0] + R[r*3+1]*qp[1] + R[r*3+2]*qp[2] + t[r];
            float kg = R[r*3+0]*kp[0] + R[r*3+1]*kp[1] + R[r*3+2]*kp[2] + t[r];
            qsq += qg * qg; ksq += kg * kg;
            Ar[16 + p * 3 + r] = PS * pw * qg;
            Br[16 + p * 3 + r] = kg;
        }
    }
    Ar[28] = -0.5f * PS * pw * qsq;
    Ar[29] = 1.0f; Ar[30] = 0.f; Ar[31] = 0.f;
    Br[28] = 1.0f;
    Br[29] = -0.5f * PS * pw * ksq;
    Br[30] = 0.f; Br[31] = 0.f;
    // transposed value matrix: VT[h][n][i]
    bf16* V = VT + (size_t)h * 48 * NN;
#pragma unroll
    for (int d = 0; d < DV; ++d) V[(size_t)d * NN + i] = f2bf(row[384 + h * DV + d]);
#pragma unroll
    for (int p = 0; p < DVP; ++p) {
        const float* vp = row + 864 + (h * DVP + p) * 3;
#pragma unroll
        for (int r = 0; r < 3; ++r) {
            float vg = R[r*3+0]*vp[0] + R[r*3+1]*vp[1] + R[r*3+2]*vp[2] + t[r];
            V[(size_t)(16 + p * 3 + r) * NN + i] = f2bf(vg);
        }
    }
#pragma unroll
    for (int n = 40; n < 48; ++n) V[(size_t)n * NN + i] = f2bf(0.f);
}

// ---------------------------------------------------------------------------
// K4: pair layernorm + pair bias (scaled). One wave per (i,j).
//     Also caches per-row mean/rstd for pass-2 re-normalization.
// ---------------------------------------------------------------------------
__global__ void pairln_kernel(float* __restrict__ PB, float* __restrict__ MU,
                              float* __restrict__ RS, const float* __restrict__ P,
                              const float* __restrict__ gp, const float* __restrict__ bp,
                              const float* __restrict__ Wpb) {
    int wave = blockIdx.x * (blockDim.x >> 5) + (threadIdx.x >> 5);
    if (wave >= NN * NN) return;
    int lane = threadIdx.x & 31;
    const float* row = P + (size_t)wave * PAIR_DIM;
    float x[4]; float s = 0.f, s2 = 0.f;
#pragma unroll
    for (int e = 0; e < 4; ++e) { x[e] = row[lane + 32 * e]; s += x[e]; s2 += x[e] * x[e]; }
#pragma unroll
    for (int m = 16; m; m >>= 1) { s += __shfl_xor(s, m, 32); s2 += __shfl_xor(s2, m, 32); }
    float mean = s * (1.f / 128.f);
    float var  = s2 * (1.f / 128.f) - mean * mean;
    float rstd = rsqrtf(var + 1e-5f);
    if (lane == 0) { MU[wave] = mean; RS[wave] = rstd; }
    float ln[4];
#pragma unroll
    for (int e = 0; e < 4; ++e) {
        int d = lane + 32 * e;
        ln[e] = (x[e] - mean) * rstd * gp[d] + bp[d];
    }
    const float PAIR_SCALE = 0.57735026918962576f;
#pragma unroll
    for (int h = 0; h < HH; ++h) {
        float p = 0.f;
#pragma unroll
        for (int e = 0; e < 4; ++e) p += ln[e] * Wpb[(lane + 32 * e) * HH + h];
#pragma unroll
        for (int m = 16; m; m >>= 1) p += __shfl_xor(p, m, 32);
        if (lane == 0) {
            int i = wave >> 9, j = wave & 511;
            PB[((size_t)h * NN + i) * NN + j] = PAIR_SCALE * p;
        }
    }
}

// ---------------------------------------------------------------------------
// K5: fused logits GEMM in full fp32 via v_wmma_f32_16x16x4_f32.
//     C init = scaled pair bias; K=32 (scalar+point terms folded).
// ---------------------------------------------------------------------------
__global__ void logits_kernel(float* __restrict__ L, const float* __restrict__ Acat,
                              const float* __restrict__ Bcat, const float* __restrict__ PB) {
    int wave = blockIdx.x * (blockDim.x >> 5) + (threadIdx.x >> 5);
    const int tilesPerHead = 32 * 32;
    int h = wave / tilesPerHead;
    if (h >= HH) return;
    int t = wave % tilesPerHead;
    int row0 = (t >> 5) << 4, col0 = (t & 31) << 4;
    int lane = threadIdx.x & 31, mr = lane & 15, half = lane >> 4;
    const float* Ah  = Acat + (size_t)h * NN * 32;
    const float* Bh  = Bcat + (size_t)h * NN * 32;
    const float* PBh = PB   + (size_t)h * NN * NN;
    v8f c;
#pragma unroll
    for (int r = 0; r < 8; ++r)
        c[r] = PBh[(size_t)(row0 + r + 8 * half) * NN + col0 + mr];
#pragma unroll
    for (int k0 = 0; k0 < 32; k0 += 4) {
        v2f a, b;
        const float* pa = Ah + (size_t)(row0 + mr) * 32 + k0 + 2 * half;
        const float* pb = Bh + (size_t)(col0 + mr) * 32 + k0 + 2 * half;
        a[0] = pa[0]; a[1] = pa[1];
        b[0] = pb[0]; b[1] = pb[1];
        c = __builtin_amdgcn_wmma_f32_16x16x4_f32(false, a, false, b, (short)0, c, false, false);
    }
    float* Lh = L + (size_t)h * NN * NN;
#pragma unroll
    for (int r = 0; r < 8; ++r)
        Lh[(size_t)(row0 + r + 8 * half) * NN + col0 + mr] = c[r];
}

// ---------------------------------------------------------------------------
// K6: fp32 softmax over j (wave per (h,i)); writes fp32 (in place) + bf16
// ---------------------------------------------------------------------------
__global__ void softmax_kernel(float* __restrict__ ATT, bf16* __restrict__ ATTB,
                               const unsigned char* __restrict__ mask) {
    int wave = blockIdx.x * (blockDim.x >> 5) + (threadIdx.x >> 5);
    if (wave >= HH * NN) return;
    int i = wave & 511;
    int lane = threadIdx.x & 31;
    float* row = ATT + (size_t)wave * NN;
    const unsigned char* mrow = mask + (size_t)i * NN;
    float v[16];
    float mx = -3.0e38f;
#pragma unroll
    for (int e = 0; e < 16; ++e) {
        int j = lane + 32 * e;
        float x = row[j];
        if (!mrow[j]) x = -3.0e38f;
        v[e] = x; mx = fmaxf(mx, x);
    }
#pragma unroll
    for (int m = 16; m; m >>= 1) mx = fmaxf(mx, __shfl_xor(mx, m, 32));
    float s = 0.f;
#pragma unroll
    for (int e = 0; e < 16; ++e) { v[e] = __expf(v[e] - mx); s += v[e]; }
#pragma unroll
    for (int m = 16; m; m >>= 1) s += __shfl_xor(s, m, 32);
    float inv = 1.f / s;
    bf16* brow = ATTB + (size_t)wave * NN;
#pragma unroll
    for (int e = 0; e < 16; ++e) {
        int j = lane + 32 * e;
        float a = v[e] * inv;
        row[j] = a; brow[j] = f2bf(a);
    }
}

// ---------------------------------------------------------------------------
// K7: res_pair = attn @ LN(p[i]) per residue i (bf16 WMMA, fp32 accum).
//     Second pass over pair_feats, streamed by the Tensor Data Mover into
//     double-buffered LDS tiles (32x128 f32), overlapped with WMMA compute.
//     Block per i: 8 waves = 8 n-tiles of 16; wave 0 drives the TDM.
// ---------------------------------------------------------------------------
__global__ void __launch_bounds__(256)
respair_kernel(float* __restrict__ RP, const bf16* __restrict__ ATTB,
               const float* __restrict__ P, const float* __restrict__ MU,
               const float* __restrict__ RS, const float* __restrict__ gp,
               const float* __restrict__ bp) {
    __shared__ float pbuf[2][32 * PAIR_DIM];   // 2 x 16 KB double buffer
    int i = blockIdx.x;                 // 512 blocks
    int wv = threadIdx.x >> 5;          // 8 waves -> n tiles covering 128 dims
    int n0 = wv * 16;
    int lane = threadIdx.x & 31, mr = lane & 15, half = lane >> 4;
    float gn = gp[n0 + mr], bn = bp[n0 + mr];
    const bf16 z = f2bf(0.f);

    unsigned long long gbase =
        (unsigned long long)(const void*)(P + (size_t)i * NN * PAIR_DIM);
    unsigned lds0 = (unsigned)(unsigned long long)(void*)&pbuf[0][0];
    unsigned lds1 = (unsigned)(unsigned long long)(void*)&pbuf[1][0];

    if (wv == 0) {
        tdm_load_tile(gbase, lds0);     // prime tile 0
    }
    v8f c = {};
    for (int kt = 0; kt < 16; ++kt) {
        if (wv == 0) {
            if (kt < 15) {
                // issue next tile into the other buffer (its readers finished
                // at the end-of-iteration barrier of kt-1), then wait for the
                // current tile (TDM ops complete in order; <=1 outstanding).
                tdm_load_tile(gbase + (unsigned long long)(kt + 1) * 32 * PAIR_DIM * 4,
                              (kt & 1) ? lds0 : lds1);
                __builtin_amdgcn_s_wait_tensorcnt(1);
            } else {
                __builtin_amdgcn_s_wait_tensorcnt(0);
            }
        }
        __syncthreads();                 // publish LDS tile kt to all waves
        const float* buf = pbuf[kt & 1];
        int k0 = kt * 32;
        v16bf a, bb;
#pragma unroll
        for (int r = 0; r < 8; ++r) {
            int ka = k0 + ((r < 4) ? 2 * r : 2 * r + 8) + 8 * half;
            if (mr < HH) {
                const bf16* pa = ATTB + ((size_t)mr * NN + i) * NN + ka;
                a[2 * r] = pa[0]; a[2 * r + 1] = pa[1];
            } else { a[2 * r] = z; a[2 * r + 1] = z; }
            int kb = ((r < 4) ? 2 * r : 2 * r + 8) + 8 * half;          // local row pair in LDS tile
            int gk = k0 + kb;                    // global j row
            float p0 = buf[kb * PAIR_DIM + n0 + mr];
            float p1 = buf[(kb + 1) * PAIR_DIM + n0 + mr];
            float l0 = (p0 - MU[i * NN + gk])     * RS[i * NN + gk]     * gn + bn;
            float l1 = (p1 - MU[i * NN + gk + 1]) * RS[i * NN + gk + 1] * gn + bn;
            bb[2 * r] = f2bf(l0); bb[2 * r + 1] = f2bf(l1);
        }
        c = __builtin_amdgcn_wmma_f32_16x16x32_bf16(false, a, false, bb, (short)0, c, false, false);
        __syncthreads();                 // tile kt fully consumed before reuse
    }
#pragma unroll
    for (int r = 0; r < 8; ++r) {
        int hrow = r + 8 * half;
        if (hrow < HH)
            RP[((size_t)hrow * NN + i) * PAIR_DIM + n0 + mr] = c[r];
    }
}

// ---------------------------------------------------------------------------
// K8: inverse rigid + point norms + assemble bf16 feature matrix [N][2112]
// ---------------------------------------------------------------------------
__global__ void post_kernel(bf16* __restrict__ F, const float* __restrict__ RV,
                            const float* __restrict__ RP, const float* __restrict__ Rm,
                            const float* __restrict__ Tr) {
    int idx = blockIdx.x * blockDim.x + threadIdx.x;
    if (idx >= NN * HH) return;
    int i = idx / HH, h = idx % HH;
    const float* R = Rm + (size_t)i * 9;
    const float* t = Tr + (size_t)i * 3;
    const float* rv = RV + ((size_t)h * NN + i) * 48;
    bf16* f = F + (size_t)i * FEAT_DIM;
#pragma unroll
    for (int d = 0; d < DV; ++d) f[h * DV + d] = f2bf(rv[d]);
#pragma unroll
    for (int p = 0; p < DVP; ++p) {
        float x = rv[16 + p * 3 + 0] - t[0];
        float y = rv[16 + p * 3 + 1] - t[1];
        float z = rv[16 + p * 3 + 2] - t[2];
        float n2 = 1e-8f;
#pragma unroll
        for (int r = 0; r < 3; ++r) {
            float inv = R[0 * 3 + r] * x + R[1 * 3 + r] * y + R[2 * 3 + r] * z;
            f[192 + h * 24 + p * 3 + r] = f2bf(inv);
            n2 += inv * inv;
        }
        f[480 + h * DVP + p] = f2bf(sqrtf(n2));
    }
    const float* rp = RP + ((size_t)h * NN + i) * PAIR_DIM;
#pragma unroll
    for (int d = 0; d < PAIR_DIM; ++d) f[576 + h * PAIR_DIM + d] = f2bf(rp[d]);
}

// ---------------------------------------------------------------------------
// Host-side orchestration
// ---------------------------------------------------------------------------
extern "C" void kernel_launch(void* const* d_in, const int* in_sizes, int n_in,
                              void* d_out, int out_size, void* d_ws, size_t ws_size,
                              hipStream_t stream) {
    const float* node_feats   = (const float*)d_in[0];
    const float* rotations    = (const float*)d_in[1];
    const float* translations = (const float*)d_in[2];
    const float* pair_feats   = (const float*)d_in[3];
    const unsigned char* mask = (const unsigned char*)d_in[4];
    const float* W_qkv        = (const float*)d_in[5];
    const float* g_node       = (const float*)d_in[6];
    const float* b_node       = (const float*)d_in[7];
    const float* g_pair       = (const float*)d_in[8];
    const float* b_pair       = (const float*)d_in[9];
    const float* W_pair_bias  = (const float*)d_in[10];
    const float* point_weights= (const float*)d_in[11];
    const float* W_out        = (const float*)d_in[12];
    const float* b_out        = (const float*)d_in[13];

    char* ws = (char*)d_ws;
    size_t off = 0;
    auto alloc = [&](size_t bytes) { char* p = ws + off; off += bytes; return p; };
    bf16*  WqkvT   = (bf16*) alloc((size_t)QKV_DIM * NODE_DIM * 2);   // 884736
    bf16*  WoutT   = (bf16*) alloc((size_t)NODE_DIM * FEAT_DIM * 2);  // 1622016
    bf16*  xln     = (bf16*) alloc((size_t)NN * NODE_DIM * 2);
    float* qkv     = (float*)alloc((size_t)NN * QKV_DIM * 4);
    float* Acat    = (float*)alloc((size_t)HH * NN * 32 * 4);
    float* Bcat    = (float*)alloc((size_t)HH * NN * 32 * 4);
    bf16*  VT      = (bf16*) alloc((size_t)HH * 48 * NN * 2);
    float* PB      = (float*)alloc((size_t)HH * NN * NN * 4);
    float* MU      = (float*)alloc((size_t)NN * NN * 4);
    float* RS      = (float*)alloc((size_t)NN * NN * 4);
    float* ATT     = (float*)alloc((size_t)HH * NN * NN * 4);
    bf16*  ATTB    = (bf16*) alloc((size_t)HH * NN * NN * 2);
    float* RESV    = (float*)alloc((size_t)HH * NN * 48 * 4);
    float* RESPAIR = (float*)alloc((size_t)HH * NN * PAIR_DIM * 4);
    bf16*  FEAT    = (bf16*) alloc((size_t)NN * FEAT_DIM * 2);
    (void)ws_size; (void)in_sizes; (void)n_in; (void)out_size;

    // K0: weight conversion (covers max(384*1152, 2112*384) elements)
    convw_kernel<<<(FEAT_DIM * NODE_DIM + 255) / 256, 256, 0, stream>>>(WqkvT, WoutT, W_qkv, W_out);

    // K1: node layernorm
    lnnode_kernel<<<NN, 128, 0, stream>>>(xln, node_feats, g_node, b_node);

    // K2: qkv = xln @ W_qkv  (M=512, N=1152, K=384)
    {
        int tiles = (NN / 16) * (QKV_DIM / 16);
        gemm_bf16_kernel<<<(tiles + 3) / 4, 128, 0, stream>>>(
            qkv, xln, WqkvT, (const float*)nullptr, NN, QKV_DIM, NODE_DIM, NODE_DIM, NODE_DIM, QKV_DIM);
    }

    // K3: rigid transforms + fused logit operands + transposed values
    prep_kernel<<<(NN * HH + 255) / 256, 256, 0, stream>>>(
        Acat, Bcat, VT, qkv, rotations, translations, point_weights);

    // K4: pair LN + pair bias (first pass over pair_feats)
    pairln_kernel<<<(NN * NN) / 8, 256, 0, stream>>>(
        PB, MU, RS, pair_feats, g_pair, b_pair, W_pair_bias);

    // K5: fused logits (fp32 WMMA), C init = scaled pair bias
    {
        int waves = HH * 32 * 32;
        logits_kernel<<<waves / 4, 128, 0, stream>>>(ATT, Acat, Bcat, PB);
    }

    // K6: softmax
    softmax_kernel<<<(HH * NN) / 8, 256, 0, stream>>>(ATT, ATTB, mask);

    // K7a: res_scalar|res_pts = attn @ [v_s|v_p] per head (M=512, N=48, K=512)
    for (int h = 0; h < HH; ++h) {
        int tiles = (NN / 16) * (48 / 16);
        gemm_bf16_kernel<<<(tiles + 3) / 4, 128, 0, stream>>>(
            RESV + (size_t)h * NN * 48,
            ATTB + (size_t)h * NN * NN,
            VT + (size_t)h * 48 * NN,
            (const float*)nullptr, NN, 48, NN, NN, NN, 48);
    }

    // K7b: res_pair = attn @ LN(p) per residue, TDM-streamed (second pass)
    respair_kernel<<<NN, 256, 0, stream>>>(RESPAIR, ATTB, pair_feats, MU, RS, g_pair, b_pair);

    // K8: inverse rigid, norms, feature concat
    post_kernel<<<(NN * HH + 255) / 256, 256, 0, stream>>>(
        FEAT, RESV, RESPAIR, rotations, translations);

    // K9: out = feat @ W_out + b_out  (M=512, N=384, K=2112)
    {
        int tiles = (NN / 16) * (NODE_DIM / 16);
        gemm_bf16_kernel<<<(tiles + 3) / 4, 128, 0, stream>>>(
            (float*)d_out, FEAT, WoutT, b_out, NN, NODE_DIM, FEAT_DIM, FEAT_DIM, FEAT_DIM, NODE_DIM);
    }
}
